// Head_16913581211725
// MI455X (gfx1250) — compile-verified, matrix operand order checked by reading, not run on hardware
//
#include <hip/hip_runtime.h>
#include <math.h>

// ---------------------------------------------------------------------------
// Single-head causal attention for MI455X (gfx1250).
//   B=16, T=2048, C=768, H=128
//   Phase 1: Q/K/V = x @ W{q,k,v} with native fp32 WMMA (16x16x4), epilogue
//            splits results into bf16 hi/lo planes (V stored transposed).
//   Phase 2: flash attention on the bf16 pipe (16x16x32) using 3-product
//            split-bf16 emulation of fp32 (rel err ~1e-5).
// ---------------------------------------------------------------------------

typedef __attribute__((ext_vector_type(2)))  float  v2f;
typedef __attribute__((ext_vector_type(8)))  float  v8f;
typedef __attribute__((ext_vector_type(8)))  __bf16 v8bf;
typedef __attribute__((ext_vector_type(16))) __bf16 v16bf;

union V16 { v16bf v; v8bf h[2]; };

#define BB    16
#define TT    2048
#define CCH   768
#define HH    128
#define XPAD  772   // 768+4 LDS row stride, conflict-free (772 % 64 == 4)
#define PPAD2 36    // 32+4  LDS row stride for 16x32 P tile, conflict-free

__device__ __forceinline__ v8f wmma_f32(v2f a, v2f b, v8f c) {
  return __builtin_amdgcn_wmma_f32_16x16x4_f32(
      false, a, false, b, (short)0, c, false, false);
}
__device__ __forceinline__ v8f wmma_bf16(v16bf a, v16bf b, v8f c) {
  return __builtin_amdgcn_wmma_f32_16x16x32_bf16(
      false, a, false, b, (short)0, c, false, false);
}

// ---------------------------------------------------------------------------
// Phase 1: QKV projection (fp32 WMMA) + bf16 hi/lo split on store.
// grid = (B*T)/16 blocks, 128 threads (4 waves, 6 output tiles each).
// ---------------------------------------------------------------------------
__global__ __launch_bounds__(128) void qkv_proj_kernel(
    const float* __restrict__ x,
    const float* __restrict__ Wq, const float* __restrict__ Wk,
    const float* __restrict__ Wv,
    __bf16* __restrict__ qh, __bf16* __restrict__ ql,
    __bf16* __restrict__ kh, __bf16* __restrict__ kl,
    __bf16* __restrict__ vth, __bf16* __restrict__ vtl)
{
  __shared__ __align__(16) float xs[16 * XPAD];

  const int tid  = threadIdx.x;
  const int wave = tid >> 5;
  const int lane = tid & 31;
  const int hgrp = lane >> 4;
  const int lid  = lane & 15;
  const int m0   = blockIdx.x * 16;

  for (int idx = tid; idx < 16 * (CCH / 4); idx += 128) {
    const int row = idx / (CCH / 4);
    const int c4  = idx % (CCH / 4);
    const float4 t =
        reinterpret_cast<const float4*>(x + (size_t)(m0 + row) * CCH)[c4];
    reinterpret_cast<float4*>(xs + row * XPAD)[c4] = t;
  }
  __syncthreads();

  const float* Wt[6];
  int          n0s[6];
#pragma unroll
  for (int i = 0; i < 6; ++i) {
    const int t   = wave * 6 + i;
    const int mat = t >> 3;
    Wt[i]  = (mat == 0) ? Wq : ((mat == 1) ? Wk : Wv);
    n0s[i] = (t & 7) * 16;
  }

  v8f acc[6];
#pragma unroll
  for (int i = 0; i < 6; ++i)
#pragma unroll
    for (int r = 0; r < 8; ++r) acc[i][r] = 0.0f;

  for (int kk = 0; kk < CCH; kk += 4) {
    const float2 af =
        *reinterpret_cast<const float2*>(xs + lid * XPAD + kk + 2 * hgrp);
    v2f a; a[0] = af.x; a[1] = af.y;
#pragma unroll
    for (int i = 0; i < 6; ++i) {
      v2f b;
      b[0] = Wt[i][(size_t)(kk + 2 * hgrp)     * HH + n0s[i] + lid];
      b[1] = Wt[i][(size_t)(kk + 2 * hgrp + 1) * HH + n0s[i] + lid];
      acc[i] = wmma_f32(a, b, acc[i]);
    }
  }

  // Split-bf16 epilogue: Q/K row-major planes, V transposed planes.
#pragma unroll
  for (int i = 0; i < 6; ++i) {
    const int t   = wave * 6 + i;
    const int mat = t >> 3;
    const int n0  = (t & 7) * 16;
    if (mat < 2) {
      __bf16* ph = (mat == 0) ? qh : kh;
      __bf16* pl = (mat == 0) ? ql : kl;
#pragma unroll
      for (int r = 0; r < 8; ++r) {
        const int   row = r + 8 * hgrp;
        const float f   = acc[i][r];
        const __bf16 hi = (__bf16)f;
        const __bf16 lo = (__bf16)(f - (float)hi);
        const size_t off = (size_t)(m0 + row) * HH + n0 + lid;
        ph[off] = hi;
        pl[off] = lo;
      }
    } else {
      // Vt[b][h][t]: lane's 8 rows are contiguous t's -> one b128 per plane.
      v8bf hv, lv;
#pragma unroll
      for (int r = 0; r < 8; ++r) {
        const float f   = acc[i][r];
        const __bf16 hi = (__bf16)f;
        hv[r] = hi;
        lv[r] = (__bf16)(f - (float)hi);
      }
      const size_t bb  = (size_t)(m0 / TT);
      const int    t0  = m0 % TT;
      const size_t off = (bb * HH + n0 + lid) * (size_t)TT + t0 + 8 * hgrp;
      *reinterpret_cast<v8bf*>(vth + off) = hv;
      *reinterpret_cast<v8bf*>(vtl + off) = lv;
    }
  }
}

// ---------------------------------------------------------------------------
// Phase 2: causal flash attention, split-bf16 on the 16x16x32 pipe.
// 1 wave per 16-query tile, 32 keys per iteration.
// grid = (B*T/16)/4 blocks, 128 threads.
// ---------------------------------------------------------------------------
__global__ __launch_bounds__(128) void attn_kernel(
    const __bf16* __restrict__ qh, const __bf16* __restrict__ ql,
    const __bf16* __restrict__ kh, const __bf16* __restrict__ kl,
    const __bf16* __restrict__ vth, const __bf16* __restrict__ vtl,
    float* __restrict__ out)
{
  __shared__ __align__(16) float pbuf[4 * 16 * PPAD2];

  const int tid  = threadIdx.x;
  const int wave = tid >> 5;
  const int lane = tid & 31;
  const int hgrp = lane >> 4;
  const int lid  = lane & 15;
  float* pbw = pbuf + wave * 16 * PPAD2;

  const int g  = blockIdx.x * 4 + wave;
  const int b  = g >> 7;       // 128 query tiles per batch
  const int qt = g & 127;
  const int q0 = qt * 16;
  const size_t bT = (size_t)b * TT;

  const float scl = 3.6084391824351615e-2f;  // 768^-0.5

  // Q tile as 4 chunk-fragments per plane (A layout: lane p -> two 8-runs).
  V16 aqh[4], aql[4];
  {
    const __bf16* qhr = qh + (bT + q0 + lid) * (size_t)HH;
    const __bf16* qlr = ql + (bT + q0 + lid) * (size_t)HH;
#pragma unroll
    for (int ch = 0; ch < 4; ++ch) {
      aqh[ch].h[0] = *reinterpret_cast<const v8bf*>(qhr + ch * 32 + 8 * hgrp);
      aqh[ch].h[1] = *reinterpret_cast<const v8bf*>(qhr + ch * 32 + 16 + 8 * hgrp);
      aql[ch].h[0] = *reinterpret_cast<const v8bf*>(qlr + ch * 32 + 8 * hgrp);
      aql[ch].h[1] = *reinterpret_cast<const v8bf*>(qlr + ch * 32 + 16 + 8 * hgrp);
    }
  }

  v8f   o[8];
  float m_[8], l_[8];
#pragma unroll
  for (int t = 0; t < 8; ++t)
#pragma unroll
    for (int r = 0; r < 8; ++r) o[t][r] = 0.0f;
#pragma unroll
  for (int r = 0; r < 8; ++r) { m_[r] = -INFINITY; l_[r] = 0.0f; }

  const int nblk = qt / 2 + 1;               // 32-key blocks (last one masked)
  for (int blk = 0; blk < nblk; ++blk) {
    const int kb0 = blk * 32;

    // --- S = Q @ K^T : two 16x16 tiles, 4 chunks x 3 split-WMMAs each ---
    v8f s0, s1;
#pragma unroll
    for (int r = 0; r < 8; ++r) { s0[r] = 0.0f; s1[r] = 0.0f; }

    const __bf16* kh0 = kh + (bT + kb0 + lid) * (size_t)HH;
    const __bf16* kl0 = kl + (bT + kb0 + lid) * (size_t)HH;
    const __bf16* kh1 = kh0 + 16 * HH;
    const __bf16* kl1 = kl0 + 16 * HH;
#pragma unroll
    for (int ch = 0; ch < 4; ++ch) {
      V16 bh, bl;
      bh.h[0] = *reinterpret_cast<const v8bf*>(kh0 + ch * 32 + 8 * hgrp);
      bh.h[1] = *reinterpret_cast<const v8bf*>(kh0 + ch * 32 + 16 + 8 * hgrp);
      bl.h[0] = *reinterpret_cast<const v8bf*>(kl0 + ch * 32 + 8 * hgrp);
      bl.h[1] = *reinterpret_cast<const v8bf*>(kl0 + ch * 32 + 16 + 8 * hgrp);
      s0 = wmma_bf16(aqh[ch].v, bh.v, s0);
      s0 = wmma_bf16(aqh[ch].v, bl.v, s0);
      s0 = wmma_bf16(aql[ch].v, bh.v, s0);

      bh.h[0] = *reinterpret_cast<const v8bf*>(kh1 + ch * 32 + 8 * hgrp);
      bh.h[1] = *reinterpret_cast<const v8bf*>(kh1 + ch * 32 + 16 + 8 * hgrp);
      bl.h[0] = *reinterpret_cast<const v8bf*>(kl1 + ch * 32 + 8 * hgrp);
      bl.h[1] = *reinterpret_cast<const v8bf*>(kl1 + ch * 32 + 16 + 8 * hgrp);
      s1 = wmma_bf16(aqh[ch].v, bh.v, s1);
      s1 = wmma_bf16(aqh[ch].v, bl.v, s1);
      s1 = wmma_bf16(aql[ch].v, bh.v, s1);
    }

    // --- mask (after scaling, masked cols set to -inf), online softmax over 32 cols ---
    const bool lastb = (blk == nblk - 1);
#pragma unroll
    for (int r = 0; r < 8; ++r) {
      const int mrow = r + 8 * hgrp;
      const int rowg = q0 + mrow;
      float v0 = s0[r] * scl;
      float v1 = s1[r] * scl;
      if (lastb) {
        if (kb0 + lid > rowg)      v0 = -INFINITY;
        if (kb0 + 16 + lid > rowg) v1 = -INFINITY;
      }
      float rm = fmaxf(v0, v1);
#pragma unroll
      for (int off = 1; off < 16; off <<= 1)
        rm = fmaxf(rm, __shfl_xor(rm, off, 32));
      const float mn = fmaxf(m_[r], rm);
      const float p0 = __expf(v0 - mn);
      const float p1 = __expf(v1 - mn);
      float rs = p0 + p1;
#pragma unroll
      for (int off = 1; off < 16; off <<= 1)
        rs += __shfl_xor(rs, off, 32);
      const float cf = __expf(m_[r] - mn);   // first block: exp(-inf)=0
      m_[r] = mn;
      l_[r] = l_[r] * cf + rs;
#pragma unroll
      for (int t = 0; t < 8; ++t) o[t][r] *= cf;
      pbw[mrow * PPAD2 + lid]      = p0;
      pbw[mrow * PPAD2 + 16 + lid] = p1;
    }

    // Same-wave LDS RAW (DS ops in-order per wave); fence vs the compiler.
    asm volatile("s_wait_dscnt 0" ::: "memory");

    // --- reload P in A layout, split into bf16 hi/lo ---
    V16 ph, pl;
    {
      const float* prow = pbw + lid * PPAD2;
      const float4 r0 = *reinterpret_cast<const float4*>(prow + 8 * hgrp);
      const float4 r1 = *reinterpret_cast<const float4*>(prow + 8 * hgrp + 4);
      const float4 r2 = *reinterpret_cast<const float4*>(prow + 16 + 8 * hgrp);
      const float4 r3 = *reinterpret_cast<const float4*>(prow + 16 + 8 * hgrp + 4);
      const float fr[16] = {r0.x, r0.y, r0.z, r0.w, r1.x, r1.y, r1.z, r1.w,
                            r2.x, r2.y, r2.z, r2.w, r3.x, r3.y, r3.z, r3.w};
#pragma unroll
      for (int j = 0; j < 16; ++j) {
        const __bf16 hi = (__bf16)fr[j];
        ph.v[j] = hi;
        pl.v[j] = (__bf16)(fr[j] - (float)hi);
      }
    }

    // --- O += P @ V : 8 n-tiles x 3 split-WMMAs, V^T gives contiguous runs --
#pragma unroll
    for (int t = 0; t < 8; ++t) {
      const size_t vrow = ((size_t)b * HH + t * 16 + lid) * (size_t)TT + kb0;
      V16 bvh, bvl;
      bvh.h[0] = *reinterpret_cast<const v8bf*>(vth + vrow + 8 * hgrp);
      bvh.h[1] = *reinterpret_cast<const v8bf*>(vth + vrow + 16 + 8 * hgrp);
      bvl.h[0] = *reinterpret_cast<const v8bf*>(vtl + vrow + 8 * hgrp);
      bvl.h[1] = *reinterpret_cast<const v8bf*>(vtl + vrow + 16 + 8 * hgrp);
      o[t] = wmma_bf16(ph.v, bvh.v, o[t]);
      o[t] = wmma_bf16(ph.v, bvl.v, o[t]);
      o[t] = wmma_bf16(pl.v, bvh.v, o[t]);
    }
  }

  // Normalize and store fp32 output.
#pragma unroll
  for (int r = 0; r < 8; ++r) {
    const float rl   = 1.0f / l_[r];
    const int   mrow = r + 8 * hgrp;
    float* op = out + (bT + q0 + mrow) * (size_t)HH;
#pragma unroll
    for (int t = 0; t < 8; ++t) op[t * 16 + lid] = o[t][r] * rl;
  }
}

// ---------------------------------------------------------------------------
extern "C" void kernel_launch(void* const* d_in, const int* in_sizes, int n_in,
                              void* d_out, int out_size, void* d_ws,
                              size_t ws_size, hipStream_t stream) {
  (void)in_sizes; (void)n_in; (void)out_size; (void)ws_size;
  const float* x  = (const float*)d_in[0];
  const float* Wq = (const float*)d_in[1];
  const float* Wk = (const float*)d_in[2];
  const float* Wv = (const float*)d_in[3];
  float* out = (float*)d_out;

  const size_t bth = (size_t)BB * TT * HH;   // 4,194,304 elements
  __bf16* base = (__bf16*)d_ws;              // 6 planes x 8 MB = 48 MB
  __bf16* qh  = base;
  __bf16* ql  = base + bth;
  __bf16* kh  = base + 2 * bth;
  __bf16* kl  = base + 3 * bth;
  __bf16* vth = base + 4 * bth;
  __bf16* vtl = base + 5 * bth;

  qkv_proj_kernel<<<(BB * TT) / 16, 128, 0, stream>>>(
      x, Wq, Wk, Wv, qh, ql, kh, kl, vth, vtl);
  attn_kernel<<<(BB * TT / 16) / 4, 128, 0, stream>>>(
      qh, ql, kh, kl, vth, vtl, out);
}
